// FastKANLinear_74646531604833
// MI455X (gfx1250) — matrix operand order repaired and checked by reading, not compiled, not run on hardware
//
#include <hip/hip_runtime.h>

// ---------------------------------------------------------------------------
// FastKAN fused layer as a single bf16 WMMA GEMM on gfx1250 (MI455X).
//   M = 4*2048 = 8192, N = OUT_F = 1024, K = IN_F*(1+GRID) = 4096
//   A[m,k]  : k-block 0 = silu(x), blocks 1..3 = sin(g*x)      (bf16, ws)
//   Wp[o,k] : k-block 0 = base_w,  blocks 1..3 = spline_w[..g] (bf16, ws)
//   out = A @ Wp^T + bias   (fp32)
// GEMM: 128x128 block tile, 8 waves (wave32), double-buffered LDS filled by
// CDNA5 async global->LDS DMA (ASYNCcnt) when available.
// ---------------------------------------------------------------------------

typedef __bf16 bf16x8  __attribute__((ext_vector_type(8)));
typedef __bf16 bf16x16 __attribute__((ext_vector_type(16)));
typedef float  f32x8   __attribute__((ext_vector_type(8)));
typedef unsigned short u16x8 __attribute__((ext_vector_type(8)));
typedef int    i32x4   __attribute__((ext_vector_type(4)));

static constexpr int IN_F  = 1024;
static constexpr int OUT_F = 1024;
static constexpr int GRIDN = 3;
static constexpr int M_DIM = 4 * 2048;           // 8192
static constexpr int N_DIM = OUT_F;              // 1024
static constexpr int K_DIM = IN_F * (1 + GRIDN); // 4096

static constexpr int KSTEP      = 32;            // K per LDS buffer
static constexpr int LDS_STRIDE = 40;            // 32 + 8 pad (bf16 units)

#if defined(__gfx1250__) && __has_builtin(__builtin_amdgcn_global_load_async_to_lds_b128)
#define USE_ASYNC_LDS 1
#else
#define USE_ASYNC_LDS 0
#endif

#if USE_ASYNC_LDS
#if __has_builtin(__builtin_amdgcn_s_wait_asynccnt)
#define WAIT_ASYNC() __builtin_amdgcn_s_wait_asynccnt(0)
#else
#define WAIT_ASYNC() asm volatile("s_wait_asynccnt 0x0" ::: "memory")
#endif
#else
#define WAIT_ASYNC() ((void)0)
#endif

static __device__ __forceinline__ unsigned short f2bf(float f) {
    union { float f; unsigned u; } v; v.f = f;
    unsigned r = v.u + 0x7FFFu + ((v.u >> 16) & 1u);   // round-to-nearest-even
    return (unsigned short)(r >> 16);
}

// ---------------------------------------------------------------------------
// Kernel 1: activation pack.  Each thread: 8 consecutive i for one row m.
// ---------------------------------------------------------------------------
__global__ __launch_bounds__(256) void pack_act(const float* __restrict__ x,
                                                unsigned short* __restrict__ A) {
    const int t  = blockIdx.x * 256 + threadIdx.x;      // 0 .. M*IN_F/8-1
    const int m  = t >> 7;                              // IN_F/8 = 128 chunks/row
    const int i0 = (t & 127) << 3;

    const float4* xp = reinterpret_cast<const float4*>(x + (size_t)m * IN_F + i0);
    float4 a = xp[0], b = xp[1];
    float xv[8] = {a.x, a.y, a.z, a.w, b.x, b.y, b.z, b.w};

    u16x8 o0, o1, o2, o3;
#pragma unroll
    for (int j = 0; j < 8; ++j) {
        float f   = xv[j];
        float sig = 1.0f / (1.0f + __expf(-f));
        o0[j] = f2bf(f * sig);
        o1[j] = f2bf(__sinf(f));
        o2[j] = f2bf(__sinf(2.0f * f));
        o3[j] = f2bf(__sinf(3.0f * f));
    }
    const size_t base = (size_t)m * K_DIM + i0;
    *reinterpret_cast<u16x8*>(A + base)            = o0;
    *reinterpret_cast<u16x8*>(A + base + IN_F)     = o1;
    *reinterpret_cast<u16x8*>(A + base + 2 * IN_F) = o2;
    *reinterpret_cast<u16x8*>(A + base + 3 * IN_F) = o3;
}

// ---------------------------------------------------------------------------
// Kernel 2: weight pack.  Wp[o, k] row-major bf16 (k contiguous).
// ---------------------------------------------------------------------------
__global__ __launch_bounds__(256) void pack_w(const float* __restrict__ base_w,
                                              const float* __restrict__ spline_w,
                                              unsigned short* __restrict__ W) {
    const int t  = blockIdx.x * 256 + threadIdx.x;      // 0 .. OUT_F*IN_F/8-1
    const int o  = t >> 7;
    const int i0 = (t & 127) << 3;

    u16x8 w0, w1, w2, w3;
#pragma unroll
    for (int j = 0; j < 8; ++j) {
        const int i = i0 + j;
        w0[j] = f2bf(base_w[(size_t)o * IN_F + i]);
        const float* sp = spline_w + ((size_t)o * IN_F + i) * GRIDN;
        w1[j] = f2bf(sp[0]);
        w2[j] = f2bf(sp[1]);
        w3[j] = f2bf(sp[2]);
    }
    const size_t base = (size_t)o * K_DIM + i0;
    *reinterpret_cast<u16x8*>(W + base)            = w0;
    *reinterpret_cast<u16x8*>(W + base + IN_F)     = w1;
    *reinterpret_cast<u16x8*>(W + base + 2 * IN_F) = w2;
    *reinterpret_cast<u16x8*>(W + base + 3 * IN_F) = w3;
}

// ---------------------------------------------------------------------------
// GEMM helpers
// ---------------------------------------------------------------------------
// Stage one 128xKSTEP bf16 tile of A and of W into LDS.
// 128 rows * 32 cols * 2B = 8KB per matrix = 512 x 16B chunks; 2 per thread.
static __device__ __forceinline__ void stage_tile(
        const unsigned short* __restrict__ gA,
        const unsigned short* __restrict__ gW,
        unsigned short* sA, unsigned short* sB,
        int rowBase, int colBase, int kb, int tid) {
#pragma unroll
    for (int p = 0; p < 2; ++p) {
        const int id = tid + p * 256;
        const int r  = id >> 2;            // 0..127
        const int c  = (id & 3) << 3;      // 0,8,16,24 (bf16 units)
        const size_t ga = (size_t)(rowBase + r) * K_DIM + kb + c;
        const size_t gb = (size_t)(colBase + r) * K_DIM + kb + c;
        const int    ls = r * LDS_STRIDE + c;
#if USE_ASYNC_LDS
        __builtin_amdgcn_global_load_async_to_lds_b128(
            (i32x4*)(gA + ga), (i32x4*)(sA + ls), 0, 0);
        __builtin_amdgcn_global_load_async_to_lds_b128(
            (i32x4*)(gW + gb), (i32x4*)(sB + ls), 0, 0);
#else
        *reinterpret_cast<uint4*>(&sA[ls]) = *reinterpret_cast<const uint4*>(gA + ga);
        *reinterpret_cast<uint4*>(&sB[ls]) = *reinterpret_cast<const uint4*>(gW + gb);
#endif
    }
}

// One KSTEP(=32) of MACs for this wave: 4x2 accumulators of 16x16.
static __device__ __forceinline__ void compute_tile(
        const unsigned short* sA, const unsigned short* sB,
        f32x8 (&acc)[4][2], int waveM, int waveN, int laneRow, int laneHalf) {
    bf16x16 afrag[4], bfrag[2];
    // A fragment (16x32 bf16): lane<16 -> K {0..7,16..23}; lane>=16 -> +8
#pragma unroll
    for (int mi = 0; mi < 4; ++mi) {
        const int r  = waveM * 64 + mi * 16 + laneRow;
        const int c0 = laneHalf * 8;
        bf16x8 lo = *reinterpret_cast<const bf16x8*>(&sA[r * LDS_STRIDE + c0]);
        bf16x8 hi = *reinterpret_cast<const bf16x8*>(&sA[r * LDS_STRIDE + c0 + 16]);
        afrag[mi] = __builtin_shufflevector(lo, hi, 0, 1, 2, 3, 4, 5, 6, 7,
                                            8, 9, 10, 11, 12, 13, 14, 15);
    }
    // B fragment (32x16 bf16): lane<16 -> K {0..7,16..23}; lane>=16 -> +8
#pragma unroll
    for (int ni = 0; ni < 2; ++ni) {
        const int r  = waveN * 32 + ni * 16 + laneRow;
        const int c0 = laneHalf * 8;
        bf16x8 lo = *reinterpret_cast<const bf16x8*>(&sB[r * LDS_STRIDE + c0]);
        bf16x8 hi = *reinterpret_cast<const bf16x8*>(&sB[r * LDS_STRIDE + c0 + 16]);
        bfrag[ni] = __builtin_shufflevector(lo, hi, 0, 1, 2, 3, 4, 5, 6, 7,
                                            8, 9, 10, 11, 12, 13, 14, 15);
    }
#pragma unroll
    for (int mi = 0; mi < 4; ++mi)
#pragma unroll
        for (int ni = 0; ni < 2; ++ni)
            acc[mi][ni] = __builtin_amdgcn_wmma_f32_16x16x32_bf16(
                false, afrag[mi], false, bfrag[ni],
                (short)0, acc[mi][ni], false, false);
}

// ---------------------------------------------------------------------------
// Kernel 3: bf16 WMMA GEMM, double-buffered LDS + async global->LDS DMA.
// ---------------------------------------------------------------------------
__global__ __launch_bounds__(256) void gemm_kan(
        const unsigned short* __restrict__ A,   // [M, K] bf16 bits
        const unsigned short* __restrict__ W,   // [N, K] bf16 bits
        const float* __restrict__ bias,         // [N]
        float* __restrict__ out) {              // [M, N]
    __shared__ unsigned short sA[2][128 * LDS_STRIDE];
    __shared__ unsigned short sB[2][128 * LDS_STRIDE];

    const int tid      = threadIdx.x;
    const int lane     = tid & 31;
    const int wave     = tid >> 5;
    const int waveM    = wave >> 2;         // 0..1 -> 64-row strip
    const int waveN    = wave & 3;          // 0..3 -> 32-col strip
    const int laneRow  = lane & 15;
    const int laneHalf = lane >> 4;

    const int rowBase = blockIdx.y * 128;   // M tile
    const int colBase = blockIdx.x * 128;   // N tile

    f32x8 acc[4][2];
#pragma unroll
    for (int mi = 0; mi < 4; ++mi)
#pragma unroll
        for (int ni = 0; ni < 2; ++ni)
            acc[mi][ni] = (f32x8){0.f, 0.f, 0.f, 0.f, 0.f, 0.f, 0.f, 0.f};

    float bsv[2];
#pragma unroll
    for (int ni = 0; ni < 2; ++ni)
        bsv[ni] = bias[colBase + waveN * 32 + ni * 16 + laneRow];

    // Prologue: fill buffer 0.
    stage_tile(A, W, sA[0], sB[0], rowBase, colBase, 0, tid);
    WAIT_ASYNC();
    __syncthreads();

    // Main loop, unrolled by two buffers.  K_DIM / (2*KSTEP) = 64 iterations.
    for (int kb = 0; kb < K_DIM; kb += 2 * KSTEP) {
        if (kb + KSTEP < K_DIM)
            stage_tile(A, W, sA[1], sB[1], rowBase, colBase, kb + KSTEP, tid);
        compute_tile(sA[0], sB[0], acc, waveM, waveN, laneRow, laneHalf);
        WAIT_ASYNC();
        __syncthreads();

        if (kb + 2 * KSTEP < K_DIM)
            stage_tile(A, W, sA[0], sB[0], rowBase, colBase, kb + 2 * KSTEP, tid);
        compute_tile(sA[1], sB[1], acc, waveM, waveN, laneRow, laneHalf);
        WAIT_ASYNC();
        __syncthreads();
    }

    // Epilogue: C layout = VGPR r, lanes 0-15 -> M=r, lanes 16-31 -> M=r+8.
#pragma unroll
    for (int mi = 0; mi < 4; ++mi) {
#pragma unroll
        for (int ni = 0; ni < 2; ++ni) {
            const int col = colBase + waveN * 32 + ni * 16 + laneRow;
#pragma unroll
            for (int r = 0; r < 8; ++r) {
                const int row = rowBase + waveM * 64 + mi * 16 + laneHalf * 8 + r;
                out[(size_t)row * N_DIM + col] = acc[mi][ni][r] + bsv[ni];
            }
        }
    }
}

// ---------------------------------------------------------------------------
// Launcher.  d_in: { x, base_w, base_b, spline_w } (all fp32). d_out: fp32.
// d_ws: A (M*K bf16 = 64MB) followed by Wp (N*K bf16 = 8MB).
// ---------------------------------------------------------------------------
extern "C" void kernel_launch(void* const* d_in, const int* in_sizes, int n_in,
                              void* d_out, int out_size, void* d_ws, size_t ws_size,
                              hipStream_t stream) {
    (void)in_sizes; (void)n_in; (void)out_size; (void)ws_size;

    const float* x        = (const float*)d_in[0];
    const float* base_w   = (const float*)d_in[1];
    const float* base_b   = (const float*)d_in[2];
    const float* spline_w = (const float*)d_in[3];
    float*       out      = (float*)d_out;

    unsigned short* A  = (unsigned short*)d_ws;
    unsigned short* Wp = A + (size_t)M_DIM * K_DIM;     // +64MB

    pack_act<<<(M_DIM * IN_F / 8) / 256, 256, 0, stream>>>(x, A);
    pack_w<<<(OUT_F * IN_F / 8) / 256, 256, 0, stream>>>(base_w, spline_w, Wp);
    gemm_kan<<<dim3(N_DIM / 128, M_DIM / 128), 256, 0, stream>>>(A, Wp, base_b, out);
}
